// PreBayesianNMS_12008728559696
// MI455X (gfx1250) — compile-verified, matrix operand order checked by reading, not run on hardware
//
#include <hip/hip_runtime.h>
#include <hip/hip_bf16.h>

typedef __attribute__((ext_vector_type(16))) _Float16 v16h;
typedef __attribute__((ext_vector_type(8)))  float    v8f;

#define NEG_INF (-__builtin_inff())

// ---------------------------------------------------------------------------
// Pass 1: convert xyxy -> xywh, precompute "area" in converted coords
// (reference treats converted coords as corners: area = (w-x)*(h-y)).
// ---------------------------------------------------------------------------
__global__ void prep_kernel(const float4* __restrict__ in_boxes,
                            float4* __restrict__ out_boxes,
                            float*  __restrict__ areas,
                            int total) {
  int i = blockIdx.x * blockDim.x + threadIdx.x;
  if (i >= total) return;
  float4 v = in_boxes[i];
  float4 o = make_float4(v.x, v.y, v.z - v.x, v.w - v.y);
  out_boxes[i] = o;
  areas[i] = (o.z - o.x) * (o.w - o.y);
}

// ---------------------------------------------------------------------------
// Pass 2: score = max softmax prob = 1 / sum(exp(x - max)) per row of 80.
// One wave handles a 16-row tile. Row-sum done on the matrix pipe:
// A = exp(x-m) packed in the 16-bit 16x32 A layout, B = ones, 3 chained
// v_wmma_f32_16x16x32_f16 cover K = 96 (80 valid, rest zero).
// Scores below CONF_THRESHOLD stored as -inf (doubles as NMS active mask).
// ---------------------------------------------------------------------------
__global__ void __launch_bounds__(256)
score_kernel(const float* __restrict__ logits,
             float* __restrict__ scores,
             int n_rows) {
  const int wave = threadIdx.x >> 5;
  const int lane = threadIdx.x & 31;
  const int tile = blockIdx.x * 8 + wave;
  const int row0 = tile * 16;
  if (row0 >= n_rows) return;               // wave-uniform: EXEC stays all-1s

  const int g = lane >> 4;                  // 0: K {0-7,16-23,..}, 1: K {8-15,24-31,..}
  const int r = lane & 15;                  // row within tile
  const float* p = logits + (size_t)(row0 + r) * 80 + g * 8;

  // Load this lane's 40 elements of its row (16B-aligned float4s).
  float x[40];
#pragma unroll
  for (int c = 0; c < 2; ++c) {
#pragma unroll
    for (int h = 0; h < 2; ++h) {
      const float4 lo = *(const float4*)(p + c * 32 + h * 16);
      const float4 hi = *(const float4*)(p + c * 32 + h * 16 + 4);
      const int base = (c * 2 + h) * 8;
      x[base + 0] = lo.x; x[base + 1] = lo.y; x[base + 2] = lo.z; x[base + 3] = lo.w;
      x[base + 4] = hi.x; x[base + 5] = hi.y; x[base + 6] = hi.z; x[base + 7] = hi.w;
    }
  }
  { // chunk 2, first half only (K 64..79)
    const float4 lo = *(const float4*)(p + 64);
    const float4 hi = *(const float4*)(p + 68);
    x[32] = lo.x; x[33] = lo.y; x[34] = lo.z; x[35] = lo.w;
    x[36] = hi.x; x[37] = hi.y; x[38] = hi.z; x[39] = hi.w;
  }

  // Exact f32 row max: lane-local then exchange with partner lane (xor 16).
  float m = x[0];
#pragma unroll
  for (int i = 1; i < 40; ++i) m = fmaxf(m, x[i]);
  m = fmaxf(m, __shfl_xor(m, 16));

  // Row sums via 3 chained WMMAs against an all-ones B.
  v16h bones;
#pragma unroll
  for (int e = 0; e < 16; ++e) bones[e] = (_Float16)1.0f;
  v8f acc = {};

#pragma unroll
  for (int cc = 0; cc < 3; ++cc) {
    v16h a;
#pragma unroll
    for (int j = 0; j < 8; ++j)
      a[j] = (_Float16)__expf(x[(cc * 2) * 8 + j] - m);
#pragma unroll
    for (int j = 0; j < 8; ++j)
      a[8 + j] = (cc < 2) ? (_Float16)__expf(x[(cc * 2 + 1) * 8 + j] - m)
                          : (_Float16)0.0f;
    acc = __builtin_amdgcn_wmma_f32_16x16x32_f16(
        /*neg_a=*/false, a, /*neg_b=*/false, bones,
        /*c_mod=*/(short)0, acc, /*reuse_a=*/false, /*reuse_b=*/false);
  }

  // D layout: VGPR v -> row v (lanes 0-15) / row v+8 (lanes 16-31).
  if (lane == 0 || lane == 16) {
    const int roff = (lane == 0) ? 0 : 8;
#pragma unroll
    for (int v = 0; v < 8; ++v) {
      float s = 1.0f / acc[v];
      scores[row0 + roff + v] = (s >= 0.05f) ? s : NEG_INF;
    }
  }
}

// ---------------------------------------------------------------------------
// 64-bit argmax key: scores are >= 0.05 (positive => IEEE bits monotonic as
// unsigned) or -inf (inactive => key 0). High word = score bits, low word =
// ~idx so max(key) picks the highest score with FIRST-index tie-breaking,
// exactly matching jnp.argmax over where(active, scores, -inf).
// ---------------------------------------------------------------------------
__device__ __forceinline__ unsigned long long nms_key(float v, int i) {
  if (v == NEG_INF) return 0ull;
  unsigned b = __float_as_uint(v);
  return ((unsigned long long)b << 32) |
         (unsigned long long)(0xFFFFFFFFu - (unsigned)i);
}

// ---------------------------------------------------------------------------
// Pass 3: greedy NMS, one 1024-thread block per image (L2-resident working
// set; 192 MB L2). Per image:
//   (a) one-time stable compaction of positive-area boxes (degenerate boxes
//       have iou==0 in the reference formula: they can't suppress nor be
//       suppressed, only self-removed on selection),
//   (b) 100 rounds: key-based global argmax over all scores, then an IoU
//       suppression sweep over the compacted list only, skipped entirely
//       when the selected box has area_r <= 0.
// ---------------------------------------------------------------------------
#define NMS_T 1024
__global__ void __launch_bounds__(NMS_T)
nms_kernel(const float4* __restrict__ boxes,
           const float*  __restrict__ areas,
           float*        __restrict__ scores,
           float4* __restrict__ pos_box,
           float*  __restrict__ pos_area,
           int*    __restrict__ pos_idx,
           int* __restrict__ sel_idx,
           int* __restrict__ sel_ok,
           int N) {
  const int b = blockIdx.x;
  const float4* bx = boxes  + (size_t)b * N;
  const float*  ar = areas  + (size_t)b * N;
  float*        sc = scores + (size_t)b * N;
  float4*       pb = pos_box  + (size_t)b * N;
  float*        pa = pos_area + (size_t)b * N;
  int*          px = pos_idx  + (size_t)b * N;

  const int tid  = threadIdx.x;
  const int lane = tid & 31;
  const int wid  = tid >> 5;

  __shared__ int                s_wsum[32];
  __shared__ unsigned long long s_key[32];
  __shared__ float              s_ref[5];
  __shared__ int                s_npos;
  __shared__ int                s_go;

  // ---- one-time stable compaction of positive-area boxes ----
  const int chunk = (N + NMS_T - 1) / NMS_T;
  const int st = tid * chunk;
  int en = st + chunk; if (en > N) en = N;
  int cnt = 0;
  for (int i = st; i < en; ++i) cnt += (ar[i] > 0.f) ? 1 : 0;

  int incl = cnt;                              // wave inclusive scan
#pragma unroll
  for (int off = 1; off < 32; off <<= 1) {
    int o = __shfl_up(incl, off);
    if (lane >= off) incl += o;
  }
  if (lane == 31) s_wsum[wid] = incl;
  __syncthreads();
  if (wid == 0) {                              // scan the 32 wave totals
    int wv = s_wsum[lane];
    int winc = wv;
#pragma unroll
    for (int off = 1; off < 32; off <<= 1) {
      int o = __shfl_up(winc, off);
      if (lane >= off) winc += o;
    }
    s_wsum[lane] = winc - wv;                  // exclusive wave offset
    if (lane == 31) s_npos = winc;             // total positive-area count
  }
  __syncthreads();
  int ofs = s_wsum[wid] + (incl - cnt);
  for (int i = st; i < en; ++i) {
    if (ar[i] > 0.f) {
      pb[ofs] = bx[i];
      pa[ofs] = ar[i];
      px[ofs] = i;
      ++ofs;
    }
  }
  __syncthreads();
  const int npos = s_npos;

  // ---- 100 greedy rounds ----
  for (int it = 0; it < 100; ++it) {
    // global argmax over all N scores (float4 stream + 64-bit keys)
    unsigned long long key = 0ull;
    const float4* sc4 = (const float4*)sc;
    const int n4 = N >> 2;
    for (int j = tid; j < n4; j += NMS_T) {
      float4 s4 = sc4[j];
      const int i0 = j << 2;
      unsigned long long k;
      k = nms_key(s4.x, i0    ); if (k > key) key = k;
      k = nms_key(s4.y, i0 + 1); if (k > key) key = k;
      k = nms_key(s4.z, i0 + 2); if (k > key) key = k;
      k = nms_key(s4.w, i0 + 3); if (k > key) key = k;
    }
    for (int i = (n4 << 2) + tid; i < N; i += NMS_T) {
      unsigned long long k = nms_key(sc[i], i);
      if (k > key) key = k;
    }
#pragma unroll
    for (int off = 16; off > 0; off >>= 1) {
      unsigned long long ok2 = __shfl_xor(key, off);
      if (ok2 > key) key = ok2;
    }
    if (lane == 0) s_key[wid] = key;
    __syncthreads();
    if (tid == 0) {
      unsigned long long kk = s_key[0];
      for (int w = 1; w < 32; ++w)
        if (s_key[w] > kk) kk = s_key[w];
      const int ok = (kk != 0ull) ? 1 : 0;
      const int fi = ok ? (int)(0xFFFFFFFFu - (unsigned)(kk & 0xFFFFFFFFull)) : 0;
      sel_idx[b * 100 + it] = fi;
      sel_ok [b * 100 + it] = ok;
      int go = 0;
      if (ok) {
        float4 rb = bx[fi];
        float area_r = ar[fi];
        s_ref[0] = rb.x; s_ref[1] = rb.y; s_ref[2] = rb.z; s_ref[3] = rb.w;
        s_ref[4] = area_r;
        sc[fi] = NEG_INF;                  // active[idx] = False
        go = (area_r > 0.f) ? 1 : 0;       // area_r<=0 => iou==0 everywhere
      }
      s_go = go;
    }
    __syncthreads();
    if (s_go) {
      const float a_r = s_ref[0], b_r = s_ref[1];
      const float c_r = s_ref[2], d_r = s_ref[3];
      const float area_r = s_ref[4];
      for (int j = tid; j < npos; j += NMS_T) {
        float4 bb = pb[j];
        float area_b = pa[j];                // > 0 by construction
        float ih = fmaxf(0.f, fminf(c_r, bb.z) - fmaxf(a_r, bb.x));
        float iw = fmaxf(0.f, fminf(d_r, bb.w) - fmaxf(b_r, bb.y));
        float inter = ih * iw;
        float uni = area_r + area_b - inter;
        if (uni > 0.f && inter > 0.5f * uni)  // iou > 0.5
          sc[px[j]] = NEG_INF;
      }
    }
    __syncthreads();
  }
}

// ---------------------------------------------------------------------------
// Pass 4: gather outputs; recompute exact f32 softmax for the <=800 selected
// rows only. d_out = boxes[8,100,4] f32 | cls_idx[8,100] i32 | cls[8,100,80] f32
// ---------------------------------------------------------------------------
__global__ void out_kernel(const float* __restrict__ logits,
                           const float4* __restrict__ boxes_conv,
                           const int* __restrict__ sel_idx,
                           const int* __restrict__ sel_ok,
                           float* __restrict__ boxes_out,
                           int*   __restrict__ cls_idx_out,
                           float* __restrict__ cls_out,
                           int N) {
  int d = blockIdx.x * blockDim.x + threadIdx.x;
  if (d >= 800) return;
  const int b   = d / 100;
  const int idx = sel_idx[d];
  const int ok  = sel_ok[d];

  float4 bo = make_float4(0.f, 0.f, 0.f, 0.f);
  if (ok) bo = boxes_conv[(size_t)b * N + idx];
  ((float4*)boxes_out)[d] = bo;

  float* co = cls_out + (size_t)d * 80;
  if (!ok) {
    cls_idx_out[d] = 0;
    for (int c = 0; c < 80; ++c) co[c] = 0.f;
    return;
  }
  const float* p = logits + ((size_t)b * N + idx) * 80;
  float m = NEG_INF; int am = 0;
  for (int c = 0; c < 80; ++c) {
    float v = p[c];
    if (v > m) { m = v; am = c; }          // strict ">" => first max index
  }
  float s = 0.f;
  for (int c = 0; c < 80; ++c) s += __expf(p[c] - m);
  float inv = 1.f / s;
  for (int c = 0; c < 80; ++c) co[c] = __expf(p[c] - m) * inv;
  cls_idx_out[d] = am;
}

// ---------------------------------------------------------------------------
extern "C" void kernel_launch(void* const* d_in, const int* in_sizes, int n_in,
                              void* d_out, int out_size, void* d_ws, size_t ws_size,
                              hipStream_t stream) {
  const float* box_in = (const float*)d_in[0];   // [8,100000,4] xyxy f32
  const float* cls_in = (const float*)d_in[1];   // [8,100000,80] logits f32
  const int B = 8, M = 100;
  const int N  = in_sizes[0] / (B * 4);
  const int BN = B * N;

  // workspace carve-up (all live regions rewritten deterministically per call)
  float*  ws_scores  = (float*)d_ws;                      // BN f32
  float4* ws_boxes   = (float4*)(ws_scores + BN);         // BN f32x4
  float*  ws_areas   = (float*)(ws_boxes + BN);           // BN f32
  float4* ws_posbox  = (float4*)(ws_areas + BN);          // BN f32x4
  float*  ws_posarea = (float*)(ws_posbox + BN);          // BN f32
  int*    ws_posidx  = (int*)(ws_posarea + BN);           // BN i32
  int*    ws_selidx  = ws_posidx + BN;                    // B*M i32
  int*    ws_selok   = ws_selidx + B * M;                 // B*M i32

  prep_kernel<<<(BN + 255) / 256, 256, 0, stream>>>(
      (const float4*)box_in, ws_boxes, ws_areas, BN);

  const int tiles = (BN + 15) / 16;            // 16 rows per wave
  score_kernel<<<(tiles + 7) / 8, 256, 0, stream>>>(cls_in, ws_scores, BN);

  nms_kernel<<<B, NMS_T, 0, stream>>>(ws_boxes, ws_areas, ws_scores,
                                      ws_posbox, ws_posarea, ws_posidx,
                                      ws_selidx, ws_selok, N);

  float* boxes_out   = (float*)d_out;                     // 3200 f32
  int*   cls_idx_out = (int*)(boxes_out + B * M * 4);     // 800 i32
  float* cls_out     = boxes_out + B * M * 4 + B * M;     // 64000 f32
  out_kernel<<<(B * M + 255) / 256, 256, 0, stream>>>(
      cls_in, ws_boxes, ws_selidx, ws_selok, boxes_out, cls_idx_out, cls_out, N);
}
